// ISGModel_52819507806454
// MI455X (gfx1250) — matrix operand order repaired and checked by reading, not compiled, not run on hardware
//
#include <hip/hip_runtime.h>
#include <stdint.h>

// ------------------------------------------------------------------
// Problem dims (fixed by the reference)
// ------------------------------------------------------------------
constexpr int kB  = 16, kNO = 512, kNR = 1024, kNA = 8, kD = 512;
constexpr int kRelRows  = kB * kNR;        // 16384
constexpr int kObjRows  = kB * kNO;        // 8192
constexpr int kOutRowsPerB = kNO + kNR + kNO; // 2048

typedef __attribute__((ext_vector_type(16))) __bf16 v16bf;
typedef __attribute__((ext_vector_type(8)))  float  v8f;
typedef __attribute__((ext_vector_type(4)))  float  v4f;

union BFU { __bf16 h; uint16_t u; };
union HL2 { uint32_t u; __bf16 h[2]; };

__device__ __forceinline__ v8f zero8() {
  v8f z = {0.f,0.f,0.f,0.f,0.f,0.f,0.f,0.f};
  return z;
}

// ------------------------------------------------------------------
// Async global->LDS copy (CDNA5 GLOBAL_LOAD_ASYNC_TO_LDS_B128, ASYNCcnt)
// with a synchronous register-copy fallback if the builtin is absent.
// Builtin signature (from hipcc diagnostic): param0 = int4 addrspace(1)*,
// param1 = int4 addrspace(3)*, then imm offset + imm cpol.
// ------------------------------------------------------------------
#if __has_builtin(__builtin_amdgcn_global_load_async_to_lds_b128)
#define USE_ASYNC_LDS 1
#endif

typedef int v4i128 __attribute__((vector_size(16)));
typedef __attribute__((address_space(1))) v4i128* gptr128;
typedef __attribute__((address_space(3))) v4i128* lptr128;

__device__ __forceinline__ void cp_b128_to_lds(const uint16_t* __restrict__ src,
                                               uint16_t* dst_lds) {
#ifdef USE_ASYNC_LDS
  __builtin_amdgcn_global_load_async_to_lds_b128((gptr128)src, (lptr128)dst_lds,
                                                 0, 0);
#else
  uint4 t = *(const uint4*)src;
  *(uint4*)dst_lds = t;
#endif
}

__device__ __forceinline__ void wait_async_all() {
#ifdef USE_ASYNC_LDS
  asm volatile("s_wait_asynccnt 0x0" ::: "memory");
#endif
}

// ------------------------------------------------------------------
// A-fragment (16x32 bf16, hi/lo split) built from an fp32 row.
// ISA layout: lanes 0-15 (M=lane): elems 0..7 -> K k0+0..7, 8..15 -> K k0+16..23
//             lanes 16-31 (M=lane-16): K k0+8..15 and k0+24..31
// ------------------------------------------------------------------
__device__ __forceinline__ void load_a_frag_f32(const float* __restrict__ arow,
                                                int k0, int half,
                                                v16bf& ahi, v16bf& alo) {
  const float* p0 = arow + k0 + half * 8;
  float x[16];
  *(v4f*)(x + 0)  = *(const v4f*)(p0 + 0);
  *(v4f*)(x + 4)  = *(const v4f*)(p0 + 4);
  *(v4f*)(x + 8)  = *(const v4f*)(p0 + 16);
  *(v4f*)(x + 12) = *(const v4f*)(p0 + 20);
#pragma unroll
  for (int e = 0; e < 16; ++e) {
    __bf16 h = (__bf16)x[e];
    __bf16 l = (__bf16)(x[e] - (float)h);
    ahi[e] = h;
    alo[e] = l;
  }
}

// Same but from LDS rows of packed {hi,lo} bf16 pairs (one u32 per element).
__device__ __forceinline__ void load_a_frag_lds(const uint32_t* __restrict__ base,
                                                int k0, int half,
                                                v16bf& ahi, v16bf& alo) {
  const uint32_t* p0 = base + k0 + half * 8;
  uint32_t w[16];
  *(uint4*)(w + 0)  = *(const uint4*)(p0 + 0);
  *(uint4*)(w + 4)  = *(const uint4*)(p0 + 4);
  *(uint4*)(w + 8)  = *(const uint4*)(p0 + 16);
  *(uint4*)(w + 12) = *(const uint4*)(p0 + 20);
#pragma unroll
  for (int e = 0; e < 16; ++e) {
    HL2 u; u.u = w[e];
    ahi[e] = u.h[0];
    alo[e] = u.h[1];
  }
}

// ------------------------------------------------------------------
// B fragments. Weight planes are pre-swizzled so one fragment's 16 bf16
// per lane are contiguous (32B): plane[((kc*32 + nc)*32 + lane)*16 + e]
// where source elem = w[kc*32 + (lane>>4)*8 + (e&7) + (e>>3)*16][nc*16 + (lane&15)].
// ------------------------------------------------------------------
__device__ __forceinline__ v16bf load_b_frag(const uint16_t* __restrict__ plane,
                                             int kc, int nc, int lane) {
  const uint16_t* p = plane + ((((size_t)kc * 32 + nc) * 32 + lane) << 4);
  return *(const v16bf*)p;
}

// From the per-block LDS stage: region r = kcl*2 + pl holds 4KB covering
// nc 0..3 in the same (nc*32+lane)*16 order as the global plane.
__device__ __forceinline__ v16bf load_b_frag_lds(const uint16_t* __restrict__ buf,
                                                 int kcl, int pl, int nc, int lane) {
  const uint16_t* p = buf + (((kcl << 1) + pl) << 11) + ((((nc << 5) + lane)) << 4);
  return *(const v16bf*)p;
}

// Stage one chunk (4 kc-steps) of B hi+lo for this block's 4 column tiles:
// 8 regions x 4KB = 32KB, 256 threads x 8 x 16B async copies.
__device__ __forceinline__ void stage_b_chunk(const uint16_t* __restrict__ bhi,
                                              const uint16_t* __restrict__ blo,
                                              int kcg0, int nc0, int tid,
                                              uint16_t* lds) {
#pragma unroll
  for (int q = 0; q < 8; ++q) {
    const uint16_t* plane = (q & 1) ? blo : bhi;
    const uint16_t* src =
        plane + (((size_t)(kcg0 + (q >> 1)) * 32 + nc0) << 9) + tid * 8;
    uint16_t* dst = lds + (q << 11) + tid * 8;
    cp_b128_to_lds(src, dst);
  }
}

// ------------------------------------------------------------------
// wmma helper: bf16x3 fp32-accurate accumulate
// ------------------------------------------------------------------
__device__ __forceinline__ v8f wmma3(v16bf ah, v16bf al, v16bf bh, v16bf bl, v8f c) {
  c = __builtin_amdgcn_wmma_f32_16x16x32_bf16(false, ah, false, bh, (short)0, c, false, false);
  c = __builtin_amdgcn_wmma_f32_16x16x32_bf16(false, ah, false, bl, (short)0, c, false, false);
  c = __builtin_amdgcn_wmma_f32_16x16x32_bf16(false, al, false, bh, (short)0, c, false, false);
  return c;
}

// ------------------------------------------------------------------
// Block-wide GEMM pipeline: all 8 waves share nc0, B is staged in LDS
// (double-buffered, async issue of chunk c+1 overlapped with WMMA on c).
// af(kc, ...) produces the wave's A fragment for compute-step kc (0-based).
// ------------------------------------------------------------------
template <typename AF>
__device__ __forceinline__ void gemm_pipeline(AF af,
                                              const uint16_t* __restrict__ bhi,
                                              const uint16_t* __restrict__ blo,
                                              int kcBase, int numKc, int nc0,
                                              int tid, int lane,
                                              uint16_t* lds, v8f acc[4]) {
  const int nch = numKc >> 2;
  stage_b_chunk(bhi, blo, kcBase, nc0, tid, lds);
  for (int c = 0; c < nch; ++c) {
    wait_async_all();     // our async writes for chunk c have landed
    __syncthreads();      // publish to all waves; also guards buffer reuse
    if (c + 1 < nch)
      stage_b_chunk(bhi, blo, kcBase + (c + 1) * 4, nc0, tid,
                    lds + ((c + 1) & 1) * 16384);
    const uint16_t* buf = lds + (c & 1) * 16384;
#pragma unroll
    for (int kcl = 0; kcl < 4; ++kcl) {
      const int kc = c * 4 + kcl;
      v16bf ah, al;
      af(kc, ah, al);
#pragma unroll
      for (int j = 0; j < 4; ++j) {
        v16bf bh = load_b_frag_lds(buf, kcl, 0, j, lane);
        v16bf bl = load_b_frag_lds(buf, kcl, 1, j, lane);
        acc[j] = wmma3(ah, al, bh, bl, acc[j]);
      }
    }
  }
}

// Non-staged variant (used by k_attr whose waves have divergent nc0).
template <typename AF>
__device__ __forceinline__ void gemm_tiles4(AF af,
                                            const uint16_t* __restrict__ bhi,
                                            const uint16_t* __restrict__ blo,
                                            int kc0, int kc1, int nc0, int lane,
                                            v8f acc[4]) {
  for (int kc = kc0; kc < kc1; ++kc) {
    v16bf ah, al;
    af(kc, ah, al);
#pragma unroll
    for (int j = 0; j < 4; ++j) {
      v16bf bh = load_b_frag(bhi, kc, nc0 + j, lane);
      v16bf bl = load_b_frag(blo, kc, nc0 + j, lane);
      acc[j] = wmma3(ah, al, bh, bl, acc[j]);
    }
  }
}

// ------------------------------------------------------------------
// 0) Weight prep: f32 [K x 512] -> fragment-swizzled bf16 hi/lo planes.
// ------------------------------------------------------------------
__global__ void __launch_bounds__(256)
k_prep_w(const float* __restrict__ src, uint16_t* __restrict__ hi,
         uint16_t* __restrict__ lo, int K) {
  int idx = blockIdx.x * 256 + threadIdx.x;
  if (idx >= K * kD) return;
  int e    = idx & 15;
  int lane = (idx >> 4) & 31;
  int rest = idx >> 9;
  int nc   = rest & 31;   // 512/16 = 32 column tiles
  int kc   = rest >> 5;
  int k = kc * 32 + (lane >> 4) * 8 + (e & 7) + (e >> 3) * 16;
  int n = nc * 16 + (lane & 15);
  float v = src[(size_t)k * kD + n];
  BFU h, l;
  h.h = (__bf16)v;
  l.h = (__bf16)(v - (float)h.h);
  hi[idx] = h.u;
  lo[idx] = l.u;
}

// ------------------------------------------------------------------
// obj_mask[b*NO+o] = (sum_d reg_feat == 0)
// ------------------------------------------------------------------
__global__ void __launch_bounds__(256)
k_objmask(const float* __restrict__ reg, int* __restrict__ mask) {
  int r = blockIdx.x * 256 + threadIdx.x;
  if (r >= kObjRows) return;
  const float* p = reg + (size_t)r * kD;
  float s = 0.f;
  for (int n = 0; n < kD; n += 4) {
    v4f v = *(const v4f*)(p + n);
    s += v.x + v.y + v.z + v.w;
  }
  mask[r] = (s == 0.f) ? 1 : 0;
}

// ------------------------------------------------------------------
// 1) rel_emb = relu(word_emb[relw] @ e2v + b)      R=16384, K=512
// ------------------------------------------------------------------
__global__ void __launch_bounds__(256)
k_embed(const float* __restrict__ word_emb, const int* __restrict__ isg_rel,
        const uint16_t* __restrict__ whi, const uint16_t* __restrict__ wlo,
        const float* __restrict__ bias, float* __restrict__ relemb) {
  __shared__ uint16_t bstage[2 * 16384];          // 64 KB double buffer
  const int tid = threadIdx.x;
  const int wave = tid >> 5, lane = tid & 31;
  const int m = lane & 15, half = lane >> 4;
  const int row0 = blockIdx.x * 128 + wave * 16;
  const int nc0  = blockIdx.y * 4;
  const int word = isg_rel[(row0 + m) * 3 + 2];
  const float* arow = word_emb + (size_t)word * kD;
  v8f acc[4] = {zero8(), zero8(), zero8(), zero8()};
  auto af = [&](int kc, v16bf& ah, v16bf& al) {
    load_a_frag_f32(arow, kc * 32, half, ah, al);
  };
  gemm_pipeline(af, whi, wlo, 0, 16, nc0, tid, lane, bstage, acc);
#pragma unroll
  for (int j = 0; j < 4; ++j) {
    const int n = (nc0 + j) * 16 + m;
    const float bn = bias[n];
#pragma unroll
    for (int i = 0; i < 8; ++i) {
      const int rr = row0 + i + 8 * half;        // C layout: M = i + 8*half
      float v = acc[j][i] + bn;
      relemb[(size_t)rr * kD + n] = v > 0.f ? v : 0.f;
    }
  }
}

// ------------------------------------------------------------------
// 2) reg_part = reg_feat @ attr_w[512:1024]  (no bias/relu) R=8192
//    (weight plane covers K=1024 -> stage kc 16..32, A is 0-based)
// ------------------------------------------------------------------
__global__ void __launch_bounds__(256)
k_regpart(const float* __restrict__ reg_feat, const uint16_t* __restrict__ whi,
          const uint16_t* __restrict__ wlo, float* __restrict__ regpart) {
  __shared__ uint16_t bstage[2 * 16384];
  const int tid = threadIdx.x;
  const int wave = tid >> 5, lane = tid & 31;
  const int m = lane & 15, half = lane >> 4;
  const int row0 = blockIdx.x * 128 + wave * 16;
  const int nc0  = blockIdx.y * 4;
  const float* arow = reg_feat + (size_t)(row0 + m) * kD;
  v8f acc[4] = {zero8(), zero8(), zero8(), zero8()};
  auto af = [&](int kc, v16bf& ah, v16bf& al) {
    load_a_frag_f32(arow, kc * 32, half, ah, al);
  };
  gemm_pipeline(af, whi, wlo, 16, 16, nc0, tid, lane, bstage, acc);
#pragma unroll
  for (int j = 0; j < 4; ++j) {
    const int n = (nc0 + j) * 16 + m;
#pragma unroll
    for (int i = 0; i < 8; ++i) {
      const int rr = row0 + i + 8 * half;
      regpart[(size_t)rr * kD + n] = acc[j][i];
    }
  }
}

// ------------------------------------------------------------------
// 3/4/5) sbj_new / obj_new / rel_new chain. K=1536 (3x512 concat).
// mode: 0 = sbj (A = [reg[sbj], reg[obj], relemb]) -> sbjnew
//       1 = obj (A = [sbjnew,   reg[obj], relemb]) -> objnew
//       2 = rel (A = [sbjnew,   objnew,   relemb]) -> d_out rel region,
//                value = (masked_relu + relemb)/2
// ------------------------------------------------------------------
__global__ void __launch_bounds__(256)
k_relchain(const float* __restrict__ reg_feat, const float* __restrict__ relemb,
           const float* __restrict__ prev0, const float* __restrict__ prev1,
           const int* __restrict__ isg_rel,
           const uint16_t* __restrict__ whi, const uint16_t* __restrict__ wlo,
           const float* __restrict__ bias, float* __restrict__ outp, int mode) {
  __shared__ uint16_t bstage[2 * 16384];
  const int tid = threadIdx.x;
  const int wave = tid >> 5, lane = tid & 31;
  const int m = lane & 15, half = lane >> 4;
  const int row0 = blockIdx.x * 128 + wave * 16;
  const int nc0  = blockIdx.y * 4;
  const int rL  = row0 + m;
  const int bL  = rL >> 10;                       // / kNR
  const int sid = isg_rel[rL * 3 + 0];
  const int oid = isg_rel[rL * 3 + 1];
  const float* s0 = (mode == 0) ? reg_feat + (size_t)(bL * kNO + sid) * kD
                                : prev0 + (size_t)rL * kD;
  const float* s1 = (mode == 2) ? prev1 + (size_t)rL * kD
                                : reg_feat + (size_t)(bL * kNO + oid) * kD;
  const float* s2 = relemb + (size_t)rL * kD;
  v8f acc[4] = {zero8(), zero8(), zero8(), zero8()};
  auto af = [&](int kc, v16bf& ah, v16bf& al) {
    const float* p; int k;
    if (kc < 16)      { p = s0; k = kc; }
    else if (kc < 32) { p = s1; k = kc - 16; }
    else              { p = s2; k = kc - 32; }
    load_a_frag_f32(p, k * 32, half, ah, al);
  };
  gemm_pipeline(af, whi, wlo, 0, 48, nc0, tid, lane, bstage, acc);

  int wv[8];
#pragma unroll
  for (int i = 0; i < 8; ++i)
    wv[i] = isg_rel[(row0 + i + 8 * half) * 3 + 2];

#pragma unroll
  for (int j = 0; j < 4; ++j) {
    const int n = (nc0 + j) * 16 + m;
    const float bn = bias[n];
#pragma unroll
    for (int i = 0; i < 8; ++i) {
      const int rr = row0 + i + 8 * half;
      float v = acc[j][i] + bn;
      v = v > 0.f ? v : 0.f;
      if (wv[i] == 1) v = 0.f;                    // rel_mask
      if (mode == 2) {
        v = (v + relemb[(size_t)rr * kD + n]) * 0.5f;
        const int b2 = rr >> 10, i2 = rr & (kNR - 1);
        outp[((size_t)b2 * kOutRowsPerB + kNO + i2) * kD + n] = v;
      } else {
        outp[(size_t)rr * kD + n] = v;
      }
    }
  }
}

// ------------------------------------------------------------------
// 6) Fused attr pipeline. One block = 64 attr-rows = 8 objects x 8 attrs.
//    Stage 1: emb = relu(word_emb[isg_att] @ e2v + b) -> LDS (packed hi/lo)
//    Stage 2: relu(emb @ attr_w[:512] + reg_part + b), in-register 8-attr
//             reduction, obj_mask, /2, store to d_out attr region.
// ------------------------------------------------------------------
__global__ void __launch_bounds__(256)
k_attr(const float* __restrict__ word_emb, const int* __restrict__ isg_att,
       const uint16_t* __restrict__ e2v_hi, const uint16_t* __restrict__ e2v_lo,
       const float* __restrict__ e2v_b,
       const uint16_t* __restrict__ at_hi, const uint16_t* __restrict__ at_lo,
       const float* __restrict__ attr_b,
       const float* __restrict__ regpart, const int* __restrict__ objmask,
       float* __restrict__ d_out) {
  __shared__ uint32_t embHL[64 * kD];             // 128 KB of the 320 KB WGP LDS
  const int wave = threadIdx.x >> 5, lane = threadIdx.x & 31;
  const int m = lane & 15, half = lane >> 4;
  const int ar0 = blockIdx.x * 64;                // first attr-row of this block

  // ---- stage 1: 32 wave-tiles of 16x64 over the 64x512 emb tile
  for (int t = wave; t < 32; t += 8) {
    const int band = t >> 3, cg = t & 7;
    const int row0 = band * 16;
    const int word = isg_att[ar0 + row0 + m];
    const float* arow = word_emb + (size_t)word * kD;
    v8f acc[4] = {zero8(), zero8(), zero8(), zero8()};
    auto af = [&](int kc, v16bf& ah, v16bf& al) {
      load_a_frag_f32(arow, kc * 32, half, ah, al);
    };
    gemm_tiles4(af, e2v_hi, e2v_lo, 0, 16, cg * 4, lane, acc);
#pragma unroll
    for (int j = 0; j < 4; ++j) {
      const int n = (cg * 4 + j) * 16 + m;
      const float bn = e2v_b[n];
#pragma unroll
      for (int i = 0; i < 8; ++i) {
        float v = acc[j][i] + bn;
        v = v > 0.f ? v : 0.f;
        HL2 u;
        u.h[0] = (__bf16)v;
        u.h[1] = (__bf16)(v - (float)u.h[0]);
        embHL[(row0 + i + 8 * half) * kD + n] = u.u;
      }
    }
  }
  __syncthreads();

  // ---- stage 2 + in-register attr reduction
  for (int t = wave; t < 32; t += 8) {
    const int band = t >> 3, cg = t & 7;
    const int row0 = band * 16;
    const uint32_t* abase = embHL + (size_t)(row0 + m) * kD;
    v8f acc[4] = {zero8(), zero8(), zero8(), zero8()};
    auto af = [&](int kc, v16bf& ah, v16bf& al) {
      load_a_frag_lds(abase, kc * 32, half, ah, al);
    };
    gemm_tiles4(af, at_hi, at_lo, 0, 16, cg * 4, lane, acc);
    // rows 0..7 of this band = all 8 attrs of object obj0 (lane half=0),
    // rows 8..15 = object obj0+1 (half=1): reduce in registers.
    const int objL = (ar0 + row0) / kNA + half;   // flat b*NO+o
    const int msk  = objmask[objL];
#pragma unroll
    for (int j = 0; j < 4; ++j) {
      const int n = (cg * 4 + j) * 16 + m;
      const float bn = attr_b[n];
      const float rp = regpart[(size_t)objL * kD + n];
      float s = 0.f;
#pragma unroll
      for (int i = 0; i < 8; ++i) {
        float v = acc[j][i] + rp + bn;
        s += v > 0.f ? v : 0.f;
      }
      const float res = msk ? 0.f : s * 0.5f;
      const int b = objL >> 9, o = objL & (kNO - 1);
      d_out[((size_t)b * kOutRowsPerB + kNO + kNR + o) * kD + n] = res;
    }
  }
}

// ------------------------------------------------------------------
// 7) init att region of d_out with reg_feat (scale applied in LN)
// ------------------------------------------------------------------
__global__ void __launch_bounds__(256)
k_initatt(const float* __restrict__ reg, float* __restrict__ out) {
  int idx = blockIdx.x * 256 + threadIdx.x;
  if (idx >= kObjRows * kD) return;
  const int per = kNO * kD;
  const int b = idx / per, rem = idx % per;
  out[(size_t)b * kOutRowsPerB * kD + rem] = reg[idx];
}

// ------------------------------------------------------------------
// 8) scatter: att[b, sbj_id] += sbj_new; att[b, obj_id] += obj_new
// ------------------------------------------------------------------
__global__ void __launch_bounds__(256)
k_scatter(const float* __restrict__ sbjnew, const float* __restrict__ objnew,
          const int* __restrict__ isg_rel, float* __restrict__ out) {
  int idx = blockIdx.x * 256 + threadIdx.x;
  if (idx >= kRelRows * kD) return;
  const int r = idx >> 9, n = idx & (kD - 1);
  const int b = r >> 10;
  const int sid = isg_rel[r * 3 + 0];
  const int oid = isg_rel[r * 3 + 1];
  const float vs = sbjnew[idx], vo = objnew[idx];
  float* base = out + (size_t)b * kOutRowsPerB * kD;
  if (vs != 0.f) unsafeAtomicAdd(base + (size_t)sid * kD + n, vs);
  if (vo != 0.f) unsafeAtomicAdd(base + (size_t)oid * kD + n, vo);
}

// ------------------------------------------------------------------
// 9) LayerNorm in place (wave32 row reduction). att rows also get /2048.
// ------------------------------------------------------------------
__global__ void __launch_bounds__(256)
k_ln(float* __restrict__ out, const float* __restrict__ g,
     const float* __restrict__ b) {
  const int wave = threadIdx.x >> 5, lane = threadIdx.x & 31;
  const int row = blockIdx.x * 8 + wave;
  const float c = ((row & (kOutRowsPerB - 1)) < kNO) ? (1.f / (2.f * kNR)) : 1.f;
  float* p = out + (size_t)row * kD;
  float x[16], s1 = 0.f, s2 = 0.f;
#pragma unroll
  for (int k = 0; k < 16; ++k) {
    const float v = c * p[k * 32 + lane];
    x[k] = v; s1 += v; s2 += v * v;
  }
#pragma unroll
  for (int off = 16; off >= 1; off >>= 1) {
    s1 += __shfl_xor(s1, off, 32);
    s2 += __shfl_xor(s2, off, 32);
  }
  const float mu  = s1 * (1.f / kD);
  const float var = s2 * (1.f / kD) - mu * mu;
  const float inv = 1.f / sqrtf(var + 1e-5f);
#pragma unroll
  for (int k = 0; k < 16; ++k) {
    const int n = k * 32 + lane;
    p[n] = (x[k] - mu) * inv * g[n] + b[n];
  }
}

// ------------------------------------------------------------------
// host side
// ------------------------------------------------------------------
extern "C" void kernel_launch(void* const* d_in, const int* in_sizes, int n_in,
                              void* d_out, int out_size, void* d_ws, size_t ws_size,
                              hipStream_t stream) {
  (void)in_sizes; (void)n_in; (void)out_size; (void)ws_size;
  const float* reg_feat = (const float*)d_in[0];
  const int*   isg_rel  = (const int*)d_in[1];
  /* d_in[2] = isg_obj, unused by reference */
  const int*   isg_att  = (const int*)d_in[3];
  const float* word_emb = (const float*)d_in[4];
  const float* e2v_w = (const float*)d_in[5];  const float* e2v_b = (const float*)d_in[6];
  const float* sbj_w = (const float*)d_in[7];  const float* sbj_b = (const float*)d_in[8];
  const float* obj_w = (const float*)d_in[9];  const float* obj_b = (const float*)d_in[10];
  const float* rel_w = (const float*)d_in[11]; const float* rel_b = (const float*)d_in[12];
  const float* attr_w = (const float*)d_in[13]; const float* attr_b = (const float*)d_in[14];
  const float* ln_g = (const float*)d_in[15];  const float* ln_b = (const float*)d_in[16];
  float* out = (float*)d_out;

  // workspace bump allocator (~124 MB total)
  char* ws = (char*)d_ws; size_t off = 0;
  auto alloc = [&](size_t bytes) -> void* {
    void* p = ws + off;
    off = (off + bytes + 255) & ~(size_t)255;
    return p;
  };
  float* relemb  = (float*)alloc((size_t)kRelRows * kD * 4);
  float* sbjnew  = (float*)alloc((size_t)kRelRows * kD * 4);
  float* objnew  = (float*)alloc((size_t)kRelRows * kD * 4);
  float* regpart = (float*)alloc((size_t)kObjRows * kD * 4);
  int*   objmask = (int*)alloc((size_t)kObjRows * 4);
  uint16_t* e2v_hi = (uint16_t*)alloc((size_t)512  * kD * 2);
  uint16_t* e2v_lo = (uint16_t*)alloc((size_t)512  * kD * 2);
  uint16_t* sbj_hi = (uint16_t*)alloc((size_t)1536 * kD * 2);
  uint16_t* sbj_lo = (uint16_t*)alloc((size_t)1536 * kD * 2);
  uint16_t* obj_hi = (uint16_t*)alloc((size_t)1536 * kD * 2);
  uint16_t* obj_lo = (uint16_t*)alloc((size_t)1536 * kD * 2);
  uint16_t* rel_hi = (uint16_t*)alloc((size_t)1536 * kD * 2);
  uint16_t* rel_lo = (uint16_t*)alloc((size_t)1536 * kD * 2);
  uint16_t* at_hi  = (uint16_t*)alloc((size_t)1024 * kD * 2);
  uint16_t* at_lo  = (uint16_t*)alloc((size_t)1024 * kD * 2);

  const dim3 blk(256);
  // weight prep (bf16 hi/lo, fragment-swizzled)
  k_prep_w<<<(512  * kD + 255) / 256, blk, 0, stream>>>(e2v_w,  e2v_hi, e2v_lo, 512);
  k_prep_w<<<(1536 * kD + 255) / 256, blk, 0, stream>>>(sbj_w,  sbj_hi, sbj_lo, 1536);
  k_prep_w<<<(1536 * kD + 255) / 256, blk, 0, stream>>>(obj_w,  obj_hi, obj_lo, 1536);
  k_prep_w<<<(1536 * kD + 255) / 256, blk, 0, stream>>>(rel_w,  rel_hi, rel_lo, 1536);
  k_prep_w<<<(1024 * kD + 255) / 256, blk, 0, stream>>>(attr_w, at_hi,  at_lo,  1024);
  k_objmask<<<kObjRows / 256, blk, 0, stream>>>(reg_feat, objmask);

  // rel_emb and reg_part (independent)
  k_embed<<<dim3(kRelRows / 128, kD / 64), blk, 0, stream>>>(
      word_emb, isg_rel, e2v_hi, e2v_lo, e2v_b, relemb);
  k_regpart<<<dim3(kObjRows / 128, kD / 64), blk, 0, stream>>>(
      reg_feat, at_hi, at_lo, regpart);

  // sequential sbj -> obj -> rel chain
  const dim3 grel(kRelRows / 128, kD / 64);
  k_relchain<<<grel, blk, 0, stream>>>(reg_feat, relemb, nullptr, nullptr, isg_rel,
                                       sbj_hi, sbj_lo, sbj_b, sbjnew, 0);
  k_relchain<<<grel, blk, 0, stream>>>(reg_feat, relemb, sbjnew, nullptr, isg_rel,
                                       obj_hi, obj_lo, obj_b, objnew, 1);
  k_relchain<<<grel, blk, 0, stream>>>(reg_feat, relemb, sbjnew, objnew, isg_rel,
                                       rel_hi, rel_lo, rel_b, out, 2);

  // fused attr pipeline (LDS-resident intermediate)
  k_attr<<<(kB * kNO * kNA) / 64, blk, 0, stream>>>(
      word_emb, isg_att, e2v_hi, e2v_lo, e2v_b, at_hi, at_lo, attr_b,
      regpart, objmask, out);

  // att region: copy + scatter-add + LN (in place)
  k_initatt<<<(kObjRows * kD + 255) / 256, blk, 0, stream>>>(reg_feat, out);
  k_scatter<<<(kRelRows * kD + 255) / 256, blk, 0, stream>>>(sbjnew, objnew, isg_rel, out);
  k_ln<<<(kB * kOutRowsPerB) / 8, blk, 0, stream>>>(out, ln_g, ln_b);
}
